// SetAttention_Linear_20486994002300
// MI455X (gfx1250) — matrix operand order repaired and checked by reading, not compiled, not run on hardware
//
#include <hip/hip_runtime.h>
#include <hip/hip_bf16.h>

typedef __attribute__((ext_vector_type(16))) _Float16 v16h;
typedef __attribute__((ext_vector_type(8)))  _Float16 v8h;
typedef __attribute__((ext_vector_type(8)))  float    v8f;

#define T_LEN 2048
#define C_DIM 1024
#define NHEAD 16
#define HSZ   64
#define NBLK  32   // T / 64
#define KMAP  4096 // HS*HS

union AFrag { v16h v; v8h h[2]; _Float16 e[16]; };

__device__ __forceinline__ v8f wmma_f16(v16h a, v16h b, v8f c) {
  // D = A(16x32) * B(32x16) + C, f32 accumulate
  return __builtin_amdgcn_wmma_f32_16x16x32_f16(false, a, false, b, (short)0, c,
                                                false, false);
}

// A fragment: row-major f16 A, lane L holds row m0+(L&15);
// halves 0..7  = K[k0 + (L>=16 ? 8 : 0) + 0..7]
// halves 8..15 = K[k0 + (L>=16 ? 8 : 0) + 16..23]
__device__ __forceinline__ v16h load_a_g(const _Float16* __restrict__ A, int ld,
                                         int m0, int k0, int lane) {
  const int lo = lane & 15, hi = lane >> 4;
  const _Float16* p = A + (size_t)(m0 + lo) * ld + k0 + hi * 8;
  AFrag f;
  f.h[0] = *(const v8h*)(p);
  f.h[1] = *(const v8h*)(p + 16);
  return f.v;
}

// B fragment from Bt = B^T row-major [N][K]: lane L holds col n0+(L&15),
// halves 0..7 = K[k0 + (L>=16 ? 8 : 0) + 0..7], halves 8..15 = +16..23
__device__ __forceinline__ v16h load_b_g(const _Float16* __restrict__ Bt, int ld,
                                         int n0, int k0, int lane) {
  const int lo = lane & 15, hi = lane >> 4;
  const _Float16* p = Bt + (size_t)(n0 + lo) * ld + k0 + hi * 8;
  AFrag f;
  f.h[0] = *(const v8h*)(p);
  f.h[1] = *(const v8h*)(p + 16);
  return f.v;
}

// ---------------- kernel 0: f32 -> f16 convert ----------------
__global__ void cvt_f16_kernel(const float* __restrict__ src,
                               _Float16* __restrict__ dst, int n) {
  for (int i = blockIdx.x * blockDim.x + threadIdx.x; i < n;
       i += gridDim.x * blockDim.x)
    dst[i] = (_Float16)src[i];
}

// ---------------- kernel 1: Y = act(Ah @ Wt^T), f16 out ----------------
// Ah [M][K] f16 row-major, Wt [N][K] f16 row-major (== B^T). act: 0=none, 1=elu+1
__global__ void gemm_f16_act_kernel(const _Float16* __restrict__ Ah,
                                    const _Float16* __restrict__ Wt,
                                    _Float16* __restrict__ Y,
                                    int M, int N, int K, int act) {
  const int lane = threadIdx.x & 31, wv = threadIdx.x >> 5;
  const int lo = lane & 15, hi = lane >> 4;
  const int nt = N >> 5;
  const int w = blockIdx.x * 8 + wv;
  const int tm = (w / nt) * 32, tn = (w % nt) * 32;
  if (tm >= M) return;

  v8f c00 = {}, c01 = {}, c10 = {}, c11 = {};
  for (int k0 = 0; k0 < K; k0 += 32) {
    if (k0 + 64 < K) {
      __builtin_prefetch(Ah + (size_t)(tm + lo) * K + k0 + 64, 0, 1);
      __builtin_prefetch(Wt + (size_t)(tn + lo) * K + k0 + 64, 0, 1);
    }
    v16h a0 = load_a_g(Ah, K, tm, k0, lane);
    v16h a1 = load_a_g(Ah, K, tm + 16, k0, lane);
    v16h b0 = load_b_g(Wt, K, tn, k0, lane);
    v16h b1 = load_b_g(Wt, K, tn + 16, k0, lane);
    c00 = wmma_f16(a0, b0, c00);
    c01 = wmma_f16(a0, b1, c01);
    c10 = wmma_f16(a1, b0, c10);
    c11 = wmma_f16(a1, b1, c11);
  }
#pragma unroll
  for (int r = 0; r < 8; ++r) {
    const int m0 = tm + r + hi * 8, m1 = m0 + 16;
    const int n0 = tn + lo, n1 = n0 + 16;
    float v00 = c00[r], v01 = c01[r], v10 = c10[r], v11 = c11[r];
    if (act == 1) {  // phi = elu(x)+1 : x>0 ? x+1 : exp(x)
      v00 = v00 > 0.f ? v00 + 1.f : __expf(v00);
      v01 = v01 > 0.f ? v01 + 1.f : __expf(v01);
      v10 = v10 > 0.f ? v10 + 1.f : __expf(v10);
      v11 = v11 > 0.f ? v11 + 1.f : __expf(v11);
    }
    Y[(size_t)m0 * N + n0] = (_Float16)v00;
    Y[(size_t)m0 * N + n1] = (_Float16)v01;
    Y[(size_t)m1 * N + n0] = (_Float16)v10;
    Y[(size_t)m1 * N + n1] = (_Float16)v11;
  }
}

// ---------------- kernel 2: per (head, block) mapped features + cumsum ------
// m[s,d'] = (k_s (x) v_s) @ Wmap^T  built on-the-fly, then in-block prefix sums
__global__ void map_cumsum_kernel(const _Float16* __restrict__ kh,
                                  const _Float16* __restrict__ vh,
                                  const _Float16* __restrict__ Wkm, // [64][4096]
                                  const _Float16* __restrict__ Wvm, // [64][4096]
                                  const float* __restrict__ bk,
                                  const float* __restrict__ bv,
                                  _Float16* __restrict__ Kset,   // [NB][NH*64]
                                  _Float16* __restrict__ VsetT,  // [NH][64][NB]
                                  _Float16* __restrict__ Ktail,  // [T][NH*64]
                                  _Float16* __restrict__ Vtail) {
  const int h = blockIdx.x, nb = blockIdx.y;
  const int lane = threadIdx.x & 31, wv = threadIdx.x >> 5;
  const int lo = lane & 15, hi = lane >> 4;
  const int mapSel = wv >> 2;          // 0 = K-map, 1 = V-map
  const int s0 = (wv & 3) * 16;        // row group inside the 64-block
  const int tb = nb * 64;
  const int hcol = h * HSZ;

  __shared__ float sm[2][64][64];      // 32 KB

  const _Float16* __restrict__ Wm = mapSel ? Wvm : Wkm;
  const int t = tb + s0 + lo;
  const _Float16* __restrict__ krow = kh + (size_t)t * C_DIM + hcol;
  const _Float16* __restrict__ vrow = vh + (size_t)t * C_DIM + hcol;

  v8f c0 = {}, c1 = {}, c2 = {}, c3 = {};
  for (int kk = 0; kk < KMAP; kk += 32) {
    const int d = kk >> 6;
    const int e0 = (kk & 63) + hi * 8;
    const _Float16 kd = krow[d];
    v8h va = *(const v8h*)(vrow + e0);
    v8h vb = *(const v8h*)(vrow + e0 + 16);
    AFrag f;
#pragma unroll
    for (int j = 0; j < 8; ++j) {
      f.e[j]     = (_Float16)(kd * va[j]);
      f.e[8 + j] = (_Float16)(kd * vb[j]);
    }
    v16h a = f.v;
    v16h b0 = load_b_g(Wm, KMAP, 0, kk, lane);
    v16h b1 = load_b_g(Wm, KMAP, 16, kk, lane);
    v16h b2 = load_b_g(Wm, KMAP, 32, kk, lane);
    v16h b3 = load_b_g(Wm, KMAP, 48, kk, lane);
    c0 = wmma_f16(a, b0, c0);
    c1 = wmma_f16(a, b1, c1);
    c2 = wmma_f16(a, b2, c2);
    c3 = wmma_f16(a, b3, c3);
  }
#pragma unroll
  for (int r = 0; r < 8; ++r) {
    const int s = s0 + r + hi * 8;
    sm[mapSel][s][lo]      = c0[r];
    sm[mapSel][s][16 + lo] = c1[r];
    sm[mapSel][s][32 + lo] = c2[r];
    sm[mapSel][s][48 + lo] = c3[r];
  }
  __syncthreads();

  // 128 threads: serial in-block prefix sum over s for one (map, d') column
  const int tid = threadIdx.x;
  if (tid < 128) {
    const int mp = tid >> 6, dc = tid & 63;
    const float bias = mp ? bv[dc] : bk[dc];
    float run = 0.f;
    for (int s = 0; s < 64; ++s) {
      run += sm[mp][s][dc];
      const float val = run + bias;
      const size_t gi = (size_t)(tb + s) * C_DIM + hcol + dc;
      if (mp == 0) Ktail[gi] = (_Float16)val;
      else         Vtail[gi] = (_Float16)val;
    }
    const float sv = run + bias;  // full-block sum == set feature
    if (mp == 0) Kset[(size_t)nb * C_DIM + hcol + dc] = (_Float16)sv;
    else         VsetT[((size_t)h * 64 + dc) * NBLK + nb] = (_Float16)sv;
  }
}

// ---------------- kernel 3: set attention (softmax over 32 sets + tail) -----
__global__ void attn_kernel(const _Float16* __restrict__ qh,
                            const _Float16* __restrict__ Kset,   // [NB][NH*64]
                            const _Float16* __restrict__ VsetT,  // [NH][64][NB]
                            const _Float16* __restrict__ Ktail,  // [T][NH*64]
                            const _Float16* __restrict__ Vtail,  // [T][NH*64]
                            _Float16* __restrict__ ctx) {        // [T][NH*64]
  const int h = blockIdx.x;
  const int lane = threadIdx.x & 31, wv = threadIdx.x >> 5;
  const int lo = lane & 15, hi = lane >> 4;
  const int t0 = blockIdx.y * 128 + wv * 16;
  const int hcol = h * HSZ;
  const float scale = 0.125f;  // 1/sqrt(64)

  __shared__ float sS[8][16][33];                     // logits (+tail at [32])
  __shared__ __align__(16) _Float16 sAtt[8][16][32];  // set probabilities
  __shared__ float sTailP[8][16];                     // tail probability

  // ---- set logits: q[16x64] @ Kset_h^T[64x32]
  v8f cs0 = {}, cs1 = {};
  for (int k0 = 0; k0 < HSZ; k0 += 32) {
    v16h a = load_a_g(qh + hcol, C_DIM, t0, k0, lane);
    v16h b0 = load_b_g(Kset + hcol, C_DIM, 0, k0, lane);
    v16h b1 = load_b_g(Kset + hcol, C_DIM, 16, k0, lane);
    cs0 = wmma_f16(a, b0, cs0);
    cs1 = wmma_f16(a, b1, cs1);
  }
  // ---- tail logit: per-row dot(q_t, Ktail_t); lane pair (lo, lo+16) splits HS
  {
    const int t = t0 + lo;
    const _Float16* __restrict__ qr = qh + (size_t)t * C_DIM + hcol + hi * 32;
    const _Float16* __restrict__ kr = Ktail + (size_t)t * C_DIM + hcol + hi * 32;
    float s = 0.f;
#pragma unroll
    for (int j = 0; j < 32; ++j) s += (float)qr[j] * (float)kr[j];
    s += __shfl_xor(s, 16, 32);
    if (hi == 0) sS[wv][lo][32] = s * scale;
  }
#pragma unroll
  for (int r = 0; r < 8; ++r) {
    const int m = r + hi * 8;
    const int t = t0 + m;
    const int blk = t >> 6;  // query's own block index
    sS[wv][m][lo]      = (lo      < blk) ? cs0[r] * scale : -1e30f;
    sS[wv][m][16 + lo] = (16 + lo < blk) ? cs1[r] * scale : -1e30f;
  }
  __syncthreads();

  // ---- softmax over 33 entries, one thread per row
  if (threadIdx.x < 128) {
    const int w2 = threadIdx.x >> 4, m = threadIdx.x & 15;
    float mx = -1e30f;
    for (int i = 0; i < 33; ++i) mx = fmaxf(mx, sS[w2][m][i]);
    float sum = 0.f;
    for (int i = 0; i < 33; ++i) sum += __expf(sS[w2][m][i] - mx);
    const float inv = 1.f / sum;
    for (int i = 0; i < 32; ++i)
      sAtt[w2][m][i] = (_Float16)(__expf(sS[w2][m][i] - mx) * inv);
    sTailP[w2][m] = __expf(sS[w2][m][32] - mx) * inv;
  }
  __syncthreads();

  // ---- out = att[16x32] @ Vset[32x64] + p_tail * Vtail
  AFrag f;
  const _Float16* arow = &sAtt[wv][lo][0];
  f.h[0] = *(const v8h*)(arow + hi * 8);
  f.h[1] = *(const v8h*)(arow + hi * 8 + 16);
  const v16h a = f.v;
  const _Float16* __restrict__ Vt = VsetT + (size_t)h * 64 * NBLK;
  v8f o0 = {}, o1 = {}, o2 = {}, o3 = {};
  o0 = wmma_f16(a, load_b_g(Vt, NBLK, 0, 0, lane), o0);
  o1 = wmma_f16(a, load_b_g(Vt, NBLK, 16, 0, lane), o1);
  o2 = wmma_f16(a, load_b_g(Vt, NBLK, 32, 0, lane), o2);
  o3 = wmma_f16(a, load_b_g(Vt, NBLK, 48, 0, lane), o3);
#pragma unroll
  for (int r = 0; r < 8; ++r) {
    const int m = r + hi * 8;
    const int t = t0 + m;
    const float pt = sTailP[wv][m];
    const size_t base = (size_t)t * C_DIM + hcol;
    ctx[base + lo]      = (_Float16)(o0[r] + pt * (float)Vtail[base + lo]);
    ctx[base + 16 + lo] = (_Float16)(o1[r] + pt * (float)Vtail[base + 16 + lo]);
    ctx[base + 32 + lo] = (_Float16)(o2[r] + pt * (float)Vtail[base + 32 + lo]);
    ctx[base + 48 + lo] = (_Float16)(o3[r] + pt * (float)Vtail[base + 48 + lo]);
  }
}

// ---------------- kernel 4: d_out = ctx @ Wc^T + bc, f32 out ----------------
__global__ void gemm_out_kernel(const _Float16* __restrict__ Ah,
                                const _Float16* __restrict__ Wt,
                                const float* __restrict__ bias,
                                float* __restrict__ Y, int M, int N, int K) {
  const int lane = threadIdx.x & 31, wv = threadIdx.x >> 5;
  const int lo = lane & 15, hi = lane >> 4;
  const int nt = N >> 5;
  const int w = blockIdx.x * 8 + wv;
  const int tm = (w / nt) * 32, tn = (w % nt) * 32;
  if (tm >= M) return;

  v8f c00 = {}, c01 = {}, c10 = {}, c11 = {};
  for (int k0 = 0; k0 < K; k0 += 32) {
    if (k0 + 64 < K) {
      __builtin_prefetch(Ah + (size_t)(tm + lo) * K + k0 + 64, 0, 1);
      __builtin_prefetch(Wt + (size_t)(tn + lo) * K + k0 + 64, 0, 1);
    }
    v16h a0 = load_a_g(Ah, K, tm, k0, lane);
    v16h a1 = load_a_g(Ah, K, tm + 16, k0, lane);
    v16h b0 = load_b_g(Wt, K, tn, k0, lane);
    v16h b1 = load_b_g(Wt, K, tn + 16, k0, lane);
    c00 = wmma_f16(a0, b0, c00);
    c01 = wmma_f16(a0, b1, c01);
    c10 = wmma_f16(a1, b0, c10);
    c11 = wmma_f16(a1, b1, c11);
  }
#pragma unroll
  for (int r = 0; r < 8; ++r) {
    const int m0 = tm + r + hi * 8, m1 = m0 + 16;
    const int n0 = tn + lo, n1 = n0 + 16;
    Y[(size_t)m0 * N + n0] = c00[r] + bias[n0];
    Y[(size_t)m0 * N + n1] = c01[r] + bias[n1];
    Y[(size_t)m1 * N + n0] = c10[r] + bias[n0];
    Y[(size_t)m1 * N + n1] = c11[r] + bias[n1];
  }
}

extern "C" void kernel_launch(void* const* d_in, const int* in_sizes, int n_in,
                              void* d_out, int out_size, void* d_ws, size_t ws_size,
                              hipStream_t stream) {
  const float* x     = (const float*)d_in[0];  // [T, C]
  const float* Wq    = (const float*)d_in[1];  // [C, C]
  const float* Wk    = (const float*)d_in[2];
  const float* Wv    = (const float*)d_in[3];
  const float* Wkmap = (const float*)d_in[4];  // [64, 4096]
  const float* bkmap = (const float*)d_in[5];  // [64]
  const float* Wvmap = (const float*)d_in[6];
  const float* bvmap = (const float*)d_in[7];
  const float* Wc    = (const float*)d_in[8];  // [C, C]
  const float* bc    = (const float*)d_in[9];  // [C]

  // ---- workspace carve-up (f16 elements)
  _Float16* w16 = (_Float16*)d_ws;
  size_t off = 0;
  _Float16* xh  = w16 + off; off += (size_t)T_LEN * C_DIM;
  _Float16* Wqh = w16 + off; off += (size_t)C_DIM * C_DIM;
  _Float16* Wkh = w16 + off; off += (size_t)C_DIM * C_DIM;
  _Float16* Wvh = w16 + off; off += (size_t)C_DIM * C_DIM;
  _Float16* Wch = w16 + off; off += (size_t)C_DIM * C_DIM;
  _Float16* Wkmh = w16 + off; off += (size_t)HSZ * KMAP;
  _Float16* Wvmh = w16 + off; off += (size_t)HSZ * KMAP;
  _Float16* qh  = w16 + off; off += (size_t)T_LEN * C_DIM;
  _Float16* kh  = w16 + off; off += (size_t)T_LEN * C_DIM;
  _Float16* vh  = w16 + off; off += (size_t)T_LEN * C_DIM;
  _Float16* Kset  = w16 + off; off += (size_t)NBLK * C_DIM;
  _Float16* VsetT = w16 + off; off += (size_t)NHEAD * HSZ * NBLK;
  _Float16* Ktail = w16 + off; off += (size_t)T_LEN * C_DIM;
  _Float16* Vtail = w16 + off; off += (size_t)T_LEN * C_DIM;
  _Float16* ctx   = w16 + off; off += (size_t)T_LEN * C_DIM;

  auto cvt = [&](const float* s, _Float16* d, int n) {
    int blocks = (n + 1023) / 1024;
    cvt_f16_kernel<<<blocks, 256, 0, stream>>>(s, d, n);
  };
  cvt(x, xh, T_LEN * C_DIM);
  cvt(Wq, Wqh, C_DIM * C_DIM);
  cvt(Wk, Wkh, C_DIM * C_DIM);
  cvt(Wv, Wvh, C_DIM * C_DIM);
  cvt(Wc, Wch, C_DIM * C_DIM);
  cvt(Wkmap, Wkmh, HSZ * KMAP);
  cvt(Wvmap, Wvmh, HSZ * KMAP);

  // projections: 2048x1024x1024; (M/32)*(N/32)=2048 wave-tiles / 8 waves = 256 blocks
  gemm_f16_act_kernel<<<256, 256, 0, stream>>>(xh, Wqh, qh, T_LEN, C_DIM, C_DIM, 1);
  gemm_f16_act_kernel<<<256, 256, 0, stream>>>(xh, Wkh, kh, T_LEN, C_DIM, C_DIM, 1);
  gemm_f16_act_kernel<<<256, 256, 0, stream>>>(xh, Wvh, vh, T_LEN, C_DIM, C_DIM, 0);

  // mapped features + in-block prefix sums: one block per (head, set-block)
  map_cumsum_kernel<<<dim3(NHEAD, NBLK), 256, 0, stream>>>(
      kh, vh, Wkmh, Wvmh, bkmap, bvmap, Kset, VsetT, Ktail, Vtail);

  // attention: one block per (head, 128 query rows)
  attn_kernel<<<dim3(NHEAD, T_LEN / 128), 256, 0, stream>>>(
      qh, Kset, VsetT, Ktail, Vtail, ctx);

  // output projection with bias -> f32 d_out
  gemm_out_kernel<<<256, 256, 0, stream>>>(ctx, Wch, bc, (float*)d_out,
                                           T_LEN, C_DIM, C_DIM);
}
